// PMHungarianMatcher_2164663517209
// MI455X (gfx1250) — matrix operand order repaired and checked by reading, not compiled, not run on hardware
//
#include <hip/hip_runtime.h>
#include <hip/hip_bf16.h>

// ---------------------------------------------------------------------------
// Hungarian matcher cost matrix for MaskFormer-style matching on gfx1250.
//
// Math reduction:  softplus(-x) - softplus(x) = -x, so
//   cost_mask[q,t] = ( sum_g softplus(x[q,g]) - (X @ M^T)[q,t] ) / G
//   cost_dice[q,t] = 1 - (2*(sig(X) @ M^T)[q,t] + 1)
//                        / (sum_g sig(x[q,g]) + sum_g M[t,g] + 1)
// Heavy part: two batched [100 x 16384] x [16384 x 50] GEMMs sharing the
// same B operand -> v_wmma_f32_16x16x32_bf16 with fp32 accumulation.
// Memory-bound: 157 MB input @ 23.3 TB/s ~= 6.7 us floor.
// ---------------------------------------------------------------------------

typedef __attribute__((ext_vector_type(16))) __bf16 v16bf;
typedef __attribute__((ext_vector_type(8)))  __bf16 v8bf;
typedef __attribute__((ext_vector_type(8)))  float  v8f;

#define B_   16
#define Q_   100
#define T_   50
#define G_   16384
#define L_   21
#define QT_  7      // ceil(100/16)
#define GC_  8      // G split into 8 chunks of 2048
#define CHUNK_ 2048
#define STAGE_ 128  // G elements per LDS stage
#define NSTG_  16   // 2048/128

// Workspace layout (floats)
#define WS_A1  0
#define WS_A2  917504          // 16*7*8*16*64
#define WS_SP  1835008         // + 917504
#define WS_SG  1849344         // + 16*7*8*16
#define WS_MS  1863680         // + 14336 ; size 16*8*64 -> total 1871872 floats (~7.5MB)

// LDS fragment slot stride: 32B data + 16B pad = 48B = 24 bf16 elements.
// Fragment loads: lane L reads 2x ds_load_b128 at slot L; 12-dword lane
// stride -> 16 lanes cover all 64 banks exactly once (conflict-free).
#define SLOT_ 24

__global__ __launch_bounds__(128) void matcher_main(
    const float* __restrict__ X,   // [B,Q,G] masks_queries_logits
    const float* __restrict__ M,   // [B,T,G] mask_labels
    float* __restrict__ ws)
{
    const int bx  = blockIdx.x;
    const int gc  = bx & 7;
    const int qt  = (bx >> 3) % 7;
    const int b   = bx / 56;
    const int tid = threadIdx.x;
    const int lane = tid & 31;
    const int wv   = tid >> 5;            // wave id == T-tile (0..3)

    // A-fragment LDS (x and sigmoid(x)), B-fragment LDS (mask labels)
    __shared__ __bf16 sXF[4 * 32 * SLOT_];          // 4 kk * 32 lanes
    __shared__ __bf16 sSF[4 * 32 * SLOT_];
    __shared__ __bf16 sMF[4 * 4 * 32 * SLOT_];      // 4 kk * 4 tt * 32 lanes

    // ---- X loader role: 16 rows x 8 threads/row, 2 runs of 8 g per stage
    const int xm = tid >> 3;              // q row within tile (0..15)
    const int xj = tid & 7;
    const int q  = qt * 16 + xm;
    const bool qvalid = (q < Q_);
    const int qc = qvalid ? q : 0;
    const float* xrow = X + (size_t)(b * Q_ + qc) * G_ + gc * CHUNK_;

    // ---- M loader role: 64 rows x 2 threads/row, 4 runs of 16 g per stage
    const int mt = tid >> 1;              // t row (0..63, padded)
    const int mj = tid & 1;
    const bool tvalid = (mt < T_);
    const int tc = tvalid ? mt : 0;
    const float* mrow = M + (size_t)(b * T_ + tc) * G_ + gc * CHUNK_;

    float sp_sum = 0.f, sg_sum = 0.f, m_sum = 0.f;
    v8f acc1 = {};   // X @ M^T
    v8f acc2 = {};   // sigmoid(X) @ M^T

    for (int st = 0; st < NSTG_; ++st) {
        __syncthreads();                  // prev compute done before overwrite
        const int gbase = st * STAGE_;

        // ---------------- X: transform + store in A-fragment order ---------
        #pragma unroll
        for (int rr = 0; rr < 2; ++rr) {
            const int goff = xj * 8 + rr * 64;   // run of 8 contiguous g
            float xs[8];
            if (qvalid) {
                const float4* p = reinterpret_cast<const float4*>(xrow + gbase + goff);
                float4 a = p[0], c = p[1];
                xs[0] = a.x; xs[1] = a.y; xs[2] = a.z; xs[3] = a.w;
                xs[4] = c.x; xs[5] = c.y; xs[6] = c.z; xs[7] = c.w;
            } else {
                #pragma unroll
                for (int i = 0; i < 8; ++i) xs[i] = 0.f;
            }
            v8bf xb, sb;
            #pragma unroll
            for (int i = 0; i < 8; ++i) {
                const float x = xs[i];
                const float e = __expf(-fabsf(x));
                float sig = (x >= 0.f) ? (1.f / (1.f + e)) : (e / (1.f + e));
                const float sp = fmaxf(x, 0.f) + log1pf(e);
                if (qvalid) { sp_sum += sp; sg_sum += sig; } else { sig = 0.f; }
                xb[i] = (__bf16)x;
                sb[i] = (__bf16)sig;
            }
            // run -> fragment slot: r in {0,8,16,24}
            const int kk   = goff >> 5;
            const int r    = goff & 31;
            const int part = (r >> 3) & 1;        // VGPRs 0-3 vs 4-7
            const int h    = (r >> 4) & 1;        // lane half
            const int off  = (kk * 32 + (xm + 16 * h)) * SLOT_ + part * 8;
            *reinterpret_cast<v8bf*>(&sXF[off]) = xb;
            *reinterpret_cast<v8bf*>(&sSF[off]) = sb;
        }

        // ---------------- M: convert + store in B-fragment order -----------
        #pragma unroll
        for (int rr = 0; rr < 4; ++rr) {
            const int goff = (mj + 2 * rr) * 16;  // run of 16 contiguous g
            v8bf m0 = {}, m1 = {};
            float ls = 0.f;
            if (tvalid) {
                const float4* p = reinterpret_cast<const float4*>(mrow + gbase + goff);
                float4 a = p[0], bb = p[1], c = p[2], d = p[3];
                float mv[16] = { a.x,a.y,a.z,a.w, bb.x,bb.y,bb.z,bb.w,
                                 c.x,c.y,c.z,c.w, d.x,d.y,d.z,d.w };
                #pragma unroll
                for (int i = 0; i < 8; ++i) { ls += mv[i];     m0[i] = (__bf16)mv[i]; }
                #pragma unroll
                for (int i = 0; i < 8; ++i) { ls += mv[i + 8]; m1[i] = (__bf16)mv[i + 8]; }
            }
            m_sum += ls;
            const int kk = goff >> 5;
            const int h  = (goff >> 4) & 1;
            const int tt = mt >> 4;
            const int off = ((kk * 4 + tt) * 32 + ((mt & 15) + 16 * h)) * SLOT_;
            *reinterpret_cast<v8bf*>(&sMF[off])     = m0;
            *reinterpret_cast<v8bf*>(&sMF[off + 8]) = m1;
        }

        __syncthreads();                  // stores visible before compute

        // ---------------- WMMA compute: 4 K-steps x 2 accumulators ---------
        #pragma unroll
        for (int kk = 0; kk < 4; ++kk) {
            const v8bf* px = reinterpret_cast<const v8bf*>(&sXF[(kk * 32 + lane) * SLOT_]);
            const v8bf* ps = reinterpret_cast<const v8bf*>(&sSF[(kk * 32 + lane) * SLOT_]);
            const v8bf* pm = reinterpret_cast<const v8bf*>(&sMF[((kk * 4 + wv) * 32 + lane) * SLOT_]);
            v16bf ax = __builtin_shufflevector(px[0], px[1],
                        0,1,2,3,4,5,6,7,8,9,10,11,12,13,14,15);
            v16bf as = __builtin_shufflevector(ps[0], ps[1],
                        0,1,2,3,4,5,6,7,8,9,10,11,12,13,14,15);
            v16bf bm = __builtin_shufflevector(pm[0], pm[1],
                        0,1,2,3,4,5,6,7,8,9,10,11,12,13,14,15);
            acc1 = __builtin_amdgcn_wmma_f32_16x16x32_bf16(
                       false, ax, false, bm, (short)0, acc1, false, false);
            acc2 = __builtin_amdgcn_wmma_f32_16x16x32_bf16(
                       false, as, false, bm, (short)0, acc2, false, false);
        }
    }

    // ---------------- epilogue: write partials to workspace ----------------
    float* A1 = ws + WS_A1;
    float* A2 = ws + WS_A2;
    float* SP = ws + WS_SP;
    float* SG = ws + WS_SG;
    float* MS = ws + WS_MS;

    const int n  = lane & 15;
    const int hh = lane >> 4;
    const size_t obase = (size_t)(((b * 7 + qt) * 8 + gc) * 16) * 64;
    #pragma unroll
    for (int r = 0; r < 8; ++r) {
        const int m = r + 8 * hh;                 // C/D layout: rows r / r+8
        A1[obase + m * 64 + wv * 16 + n] = acc1[r];
        A2[obase + m * 64 + wv * 16 + n] = acc2[r];
    }

    // row-sum reductions: fixed order, no float atomics (determinism)
    sp_sum += __shfl_xor(sp_sum, 1, 32);
    sp_sum += __shfl_xor(sp_sum, 2, 32);
    sp_sum += __shfl_xor(sp_sum, 4, 32);
    sg_sum += __shfl_xor(sg_sum, 1, 32);
    sg_sum += __shfl_xor(sg_sum, 2, 32);
    sg_sum += __shfl_xor(sg_sum, 4, 32);
    if (xj == 0) {
        const int ridx = ((b * 7 + qt) * 8 + gc) * 16 + xm;
        SP[ridx] = sp_sum;
        SG[ridx] = sg_sum;
    }
    m_sum += __shfl_xor(m_sum, 1, 32);
    if (mj == 0 && qt == 0)                       // one writer per (b,gc,t)
        MS[(b * 8 + gc) * 64 + mt] = m_sum;
}

__global__ __launch_bounds__(64) void matcher_combine(
    const float* __restrict__ cls,   // [B,Q,L] class_queries_logits
    const int*   __restrict__ lab,   // [B,T]   class_labels
    const float* __restrict__ ws,
    float* __restrict__ out)         // [B,Q,T]
{
    const int b  = blockIdx.x / Q_;
    const int q  = blockIdx.x % Q_;
    const int qt = q >> 4;
    const int m  = q & 15;
    const int t  = threadIdx.x;
    if (t >= T_) return;

    const float* A1 = ws + WS_A1;
    const float* A2 = ws + WS_A2;
    const float* SP = ws + WS_SP;
    const float* SG = ws + WS_SG;
    const float* MS = ws + WS_MS;

    float a1 = 0.f, a2 = 0.f, sp = 0.f, sg = 0.f, ms = 0.f;
    #pragma unroll
    for (int gc = 0; gc < GC_; ++gc) {
        const size_t base = (size_t)(((b * 7 + qt) * 8 + gc) * 16) * 64;
        a1 += A1[base + m * 64 + t];
        a2 += A2[base + m * 64 + t];
        const int ridx = ((b * 7 + qt) * 8 + gc) * 16 + m;
        sp += SP[ridx];
        sg += SG[ridx];
        ms += MS[(b * 8 + gc) * 64 + t];
    }

    // classification cost: -softmax(class_logits)[label_t]
    const float* cl = cls + (size_t)(b * Q_ + q) * L_;
    float mx = cl[0];
    #pragma unroll
    for (int l = 1; l < L_; ++l) mx = fmaxf(mx, cl[l]);
    float den = 0.f;
    #pragma unroll
    for (int l = 0; l < L_; ++l) den += __expf(cl[l] - mx);
    const int lb = lab[b * T_ + t];
    const float cost_class = -__expf(cl[lb] - mx) / den;

    const float cost_mask = (sp - a1) * (1.0f / (float)G_);
    const float cost_dice = 1.0f - (2.0f * a2 + 1.0f) / (sg + ms + 1.0f);

    out[((size_t)(b * Q_ + q)) * T_ + t] = cost_mask + cost_class + cost_dice;
}

extern "C" void kernel_launch(void* const* d_in, const int* in_sizes, int n_in,
                              void* d_out, int out_size, void* d_ws, size_t ws_size,
                              hipStream_t stream) {
    const float* X   = (const float*)d_in[0];   // masks_queries_logits [B,Q,G]
    const float* cls = (const float*)d_in[1];   // class_queries_logits [B,Q,L]
    const float* M   = (const float*)d_in[2];   // mask_labels          [B,T,G]
    const int*   lb  = (const int*)d_in[3];     // class_labels         [B,T]
    float* ws  = (float*)d_ws;
    float* out = (float*)d_out;

    matcher_main<<<dim3(B_ * QT_ * GC_), dim3(128), 0, stream>>>(X, M, ws);
    matcher_combine<<<dim3(B_ * Q_), dim3(64), 0, stream>>>(cls, lb, ws, out);
    (void)in_sizes; (void)n_in; (void)out_size; (void)ws_size;
}
